// TunableQuantLinear_61074434949696
// MI455X (gfx1250) — matrix operand order repaired and checked by reading, not compiled, not run on hardware
//
#include <hip/hip_runtime.h>
#include <hip/hip_bf16.h>

typedef __attribute__((ext_vector_type(16))) _Float16     v16h;
typedef __attribute__((ext_vector_type(2)))  _Float16     h2;
typedef __attribute__((ext_vector_type(8)))  float        v8f;
typedef __attribute__((ext_vector_type(4)))  unsigned int v4u;

#define M_TOT  256          // B*S
#define K_MAIN 4096         // I
#define R_DIM  64           // low-rank outlier width
#define K_TOT  (K_MAIN + R_DIM)   // 4160, multiple of 32
#define N_TOT  11008        // O
#define MT_PER_WAVE 4       // M-tiles per wave: one dequantized B feeds 4 WMMAs

// ---------------------------------------------------------------------------
// Prep: A_ws[m, 0:4096] = (f16)x[m,:], A_ws[m, 4096:4160] = (f16)outlier_x[m,:]
// 2 MiB result stays L2-resident for the whole GEMM; folds the rank-64 outlier
// term into the main GEMM as a K-extension.
// ---------------------------------------------------------------------------
__global__ void pack_act_f16(const float* __restrict__ x,
                             const float* __restrict__ ox,
                             _Float16* __restrict__ a) {
  int idx = blockIdx.x * blockDim.x + threadIdx.x;
  if (idx >= M_TOT * K_TOT) return;
  int m = idx / K_TOT;
  int k = idx - m * K_TOT;
  float v = (k < K_MAIN) ? x[m * K_MAIN + k] : ox[m * R_DIM + (k - K_MAIN)];
  a[idx] = (_Float16)v;
}

// Outlier weights pre-divided by scale so the single epilogue FMA
// (acc*scale + bias) is correct for both the quant part and the outlier part.
__global__ void prescale_outlier(const float* __restrict__ wo,
                                 const float* __restrict__ scales,
                                 float* __restrict__ wos) {
  int idx = blockIdx.x * blockDim.x + threadIdx.x;
  if (idx >= N_TOT * R_DIM) return;
  int o = idx / R_DIM;
  wos[idx] = wo[idx] / scales[o];
}

// ---------------------------------------------------------------------------
// int4-dequant GEMM. Each wave: one 16-wide N strip x 4 M-tiles.
// Software-pipelined: loads for k-step i+1 are issued before the WMMAs of
// step i and carried in loop-carried double buffers, so A/B loads stay in
// flight across the matrix ops instead of serializing on s_wait_loadcnt 0.
// ---------------------------------------------------------------------------
struct AF { union { v16h h; v4u u[2]; }; };

__global__ void __launch_bounds__(256)
qgemm_wmma_f16(const _Float16* __restrict__ A,      // [256, 4160] f16
               const int*      __restrict__ qw,     // [N, 2048] one byte per int32
               const float*    __restrict__ scales, // [N]
               const float*    __restrict__ wos,    // [N, 64] f32 (pre-divided)
               const float*    __restrict__ bias,   // [N]
               float*          __restrict__ out)    // [256, N]
{
  const int lane = threadIdx.x & 31;
  const int wave = threadIdx.x >> 5;
  const int l16  = lane & 15;
  const int half = lane >> 4;                    // 0: lanes 0-15, 1: lanes 16-31

  const int ntile  = blockIdx.x * 8 + wave;      // 0..687
  const int mtbase = blockIdx.y * MT_PER_WAVE;   // 0,4,8,12
  const int o = ntile * 16 + l16;                // weight row / output column

  const float scale = scales[o];

  // A base for this lane: row l16 of M-tile mtbase, K offset half*16.
  // Per-M-tile displacement 16*K_TOT*2 = 133120 B fits the 24-bit IOFFSET.
  const _Float16* __restrict__ abase =
      A + (size_t)(mtbase * 16 + l16) * K_TOT + half * 16;
  const int* __restrict__ qrow = qw + (size_t)o * (K_MAIN / 2) + half * 8;

  v8f acc[MT_PER_WAVE];
#pragma unroll
  for (int mt = 0; mt < MT_PER_WAVE; ++mt) acc[mt] = (v8f){};

  const h2 ksub = (h2)(_Float16)(-1032.0f);      // removes the 0x6400|(^8) bias

  // ---- pipeline prologue: loads for kb = 0 ----
  unsigned int qa[8];
  *(v4u*)(qa)     = *(const v4u*)(qrow);
  *(v4u*)(qa + 4) = *(const v4u*)(qrow + 4);
  AF af[MT_PER_WAVE];
#pragma unroll
  for (int mt = 0; mt < MT_PER_WAVE; ++mt) {
    af[mt].u[0] = *(const v4u*)(abase + (size_t)mt * 16 * K_TOT);
    af[mt].u[1] = *(const v4u*)(abase + (size_t)mt * 16 * K_TOT + 8);
  }

  // ---- main quantized K loop: 128 steps of 32, double-buffered ----
#pragma unroll 2
  for (int kb = 0; kb < K_MAIN; kb += 32) {
    // next k-step (clamped on the final iteration: harmless re-load, no OOB)
    const int kn = (kb + 32 < K_MAIN) ? (kb + 32) : kb;

    // issue next-iteration loads FIRST; they stay in flight across the WMMAs
    const int* qp = qrow + (kn >> 1);
    unsigned int qn[8];
    *(v4u*)(qn)     = *(const v4u*)(qp);
    *(v4u*)(qn + 4) = *(const v4u*)(qp + 4);
    __builtin_prefetch(qp + 512, 0, 1);   // weight stream ~2KB ahead (no reuse)

    AF an[MT_PER_WAVE];
#pragma unroll
    for (int mt = 0; mt < MT_PER_WAVE; ++mt) {
      an[mt].u[0] = *(const v4u*)(abase + (size_t)mt * 16 * K_TOT + kn);
      an[mt].u[1] = *(const v4u*)(abase + (size_t)mt * 16 * K_TOT + kn + 8);
    }

    // dequant CURRENT B (integer ALU + one v_pk_add_f16 per element pair)
    union { v16h h; unsigned int u[8]; } bf;
#pragma unroll
    for (int j = 0; j < 8; ++j) {
      unsigned int t = qa[j] ^ 0x88u;  // flip int4 sign bits of both nibbles
      unsigned int p = 0x64006400u | (t & 0xFu) | ((t << 12) & 0x000F0000u);
      union { unsigned int u; h2 h; } c;
      c.u = p;
      c.h = c.h + ksub;                // exact signed int4 pair as f16
      bf.u[j] = c.u;
    }

    // 4 back-to-back WMMAs on CURRENT fragments (distinct accumulators,
    // shared read-only B; next-iter buffers an[]/qn[] are live here, which
    // forces distinct registers and keeps their loads overlapped).
#pragma unroll
    for (int mt = 0; mt < MT_PER_WAVE; ++mt)
      acc[mt] = __builtin_amdgcn_wmma_f32_16x16x32_f16(
          false, af[mt].h, false, bf.h, (short)0, acc[mt], false, false);

    // rotate pipeline buffers (register renames after unrolling)
#pragma unroll
    for (int j = 0; j < 8; ++j) qa[j] = qn[j];
#pragma unroll
    for (int mt = 0; mt < MT_PER_WAVE; ++mt) af[mt] = an[mt];
  }

  // ---- low-rank outlier tail: 2 steps of 32 over appended K range ----
  const float* __restrict__ worow = wos + (size_t)o * R_DIM + half * 16;
#pragma unroll
  for (int kb = 0; kb < R_DIM; kb += 32) {
    AF at[MT_PER_WAVE];
#pragma unroll
    for (int mt = 0; mt < MT_PER_WAVE; ++mt) {
      at[mt].u[0] = *(const v4u*)(abase + (size_t)mt * 16 * K_TOT + K_MAIN + kb);
      at[mt].u[1] = *(const v4u*)(abase + (size_t)mt * 16 * K_TOT + K_MAIN + kb + 8);
    }
    union { v16h h; _Float16 e[16]; } bf;
    const float* wp = worow + kb;
#pragma unroll
    for (int j = 0; j < 16; ++j) bf.e[j] = (_Float16)wp[j];
#pragma unroll
    for (int mt = 0; mt < MT_PER_WAVE; ++mt)
      acc[mt] = __builtin_amdgcn_wmma_f32_16x16x32_f16(
          false, at[mt].h, false, bf.h, (short)0, acc[mt], false, false);
  }

  // ---- epilogue: C layout => row = mt*16 + half*8 + r, col = o ----
  const float bv = bias[o];
#pragma unroll
  for (int mt = 0; mt < MT_PER_WAVE; ++mt) {
    float* __restrict__ orow =
        out + (size_t)((mtbase + mt) * 16 + half * 8) * N_TOT + o;
#pragma unroll
    for (int r = 0; r < 8; ++r)
      orow[(size_t)r * N_TOT] = fmaf(acc[mt][r], scale, bv);
  }
}

// ---------------------------------------------------------------------------
extern "C" void kernel_launch(void* const* d_in, const int* in_sizes, int n_in,
                              void* d_out, int out_size, void* d_ws, size_t ws_size,
                              hipStream_t stream) {
  const float* x      = (const float*)d_in[0];  // [16,16,4096]
  const float* ox     = (const float*)d_in[1];  // [16,16,64]
  const float* wscale = (const float*)d_in[2];  // [11008,1]
  const int*   qw     = (const int*)  d_in[3];  // [11008,2048]
  const float* wo     = (const float*)d_in[4];  // [11008,64]
  const float* bias   = (const float*)d_in[5];  // [11008]
  float* out = (float*)d_out;                   // [16,16,11008]

  // workspace layout: [A_f16: 256*4160*2 B][wo_scaled: 11008*64*4 B]
  _Float16* aws = (_Float16*)d_ws;
  float* wos = (float*)((char*)d_ws + (size_t)M_TOT * K_TOT * sizeof(_Float16));

  const int totalA = M_TOT * K_TOT;
  pack_act_f16<<<(totalA + 255) / 256, 256, 0, stream>>>(x, ox, aws);

  const int totalW = N_TOT * R_DIM;
  prescale_outlier<<<(totalW + 255) / 256, 256, 0, stream>>>(wo, wscale, wos);

  dim3 grid(N_TOT / (16 * 8), (M_TOT / 16) / MT_PER_WAVE);  // (86, 4)
  qgemm_wmma_f16<<<grid, dim3(256), 0, stream>>>(aws, qw, wscale, wos, bias, out);
}
